// PaiNN_16217796510150
// MI455X (gfx1250) — matrix operand order repaired and checked by reading, not compiled, not run on hardware
//
#include <hip/hip_runtime.h>
#include <math.h>

// ---------------------------------------------------------------------------
// PaiNN on MI455X (gfx1250). Dense GEMMs via v_wmma_f32_16x16x32_f16.
// Edge aggregation is deterministic (edge_dst is sorted by construction;
// each node block reduces its contiguous edge range sequentially).
// ---------------------------------------------------------------------------

#define F_DIM   128
#define NRBF    20
#define RC      5.0f
#define NLAYERS 3

typedef __attribute__((ext_vector_type(16))) _Float16 v16h;
typedef __attribute__((ext_vector_type(8)))  float    v8f;

__device__ __forceinline__ float silu_f(float x) {
    return x / (1.0f + __expf(-x));
}

// ---------------------------------------------------------------------------
// WMMA GEMM: C[M,Nc] = act(A[M,K] @ B[K,Nc] + bias)
// A, B row-major fp32, converted to f16. Requirements (guaranteed by host):
//   M % 16 == 0, K % 32 == 0, K <= 256, Nc % 32 == 0.
// Block = 128 threads (4 waves). Block tile = 64 rows x 32 cols:
//   each wave owns 16 rows and both 16-col halves (2 accumulators, A reused).
// B stripe (K x 32) is staged once per block into LDS, pre-converted to f16,
// stored in *fragment order* so each lane's v16h is a contiguous 32B run.
// Fragment layouts used here (wave32; C per CDNA5 ISA 7.12.2):
//   A 16x32 f16 : row = lane&15; K = (lane>=16?16:0) + i (contiguous k)
//                 => per lane: one contiguous 16-float run -> float4 loads
//   B 32x16 f16 : col = lane&15; lane half selects K base 0/16; K = base + h (same contiguous k as A, not ISA table order)
//   C 16x16 f32 : VGPR r -> M = r + (lane>=16 ? 8 : 0), N = lane&15
// ---------------------------------------------------------------------------
template <int SILU>
__global__ __launch_bounds__(128)
void wmma_gemm_kernel(const float* __restrict__ A, const float* __restrict__ B,
                      const float* __restrict__ bias, float* __restrict__ C,
                      int M, int K, int Nc) {
    __shared__ _Float16 b_sh[256 * 32];     // 16 KB, frag-ordered B stripe

    const int wave  = threadIdx.x >> 5;
    const int lane  = threadIdx.x & 31;
    const int tileN = blockIdx.x * 32;
    const int tileM = (blockIdx.y * 4 + wave) * 16;
    const bool active = (tileM < M);

    // ---- stage B stripe to LDS in fragment order ----
    // element (k, col) -> b_sh[(k>>4)*512 + col*16 + (k&15)]
    for (int idx = threadIdx.x; idx < K * 32; idx += 128) {
        const int k   = idx >> 5;
        const int col = idx & 31;
        b_sh[((k >> 4) << 9) + (col << 4) + (k & 15)] =
            (_Float16)B[(size_t)k * Nc + tileN + col];
    }
    __syncthreads();

    if (active) {
        const bool hi  = (lane >= 16);
        const int  l16 = lane & 15;
        const int  rowA = tileM + l16;
        const int  kA   = hi ? 16 : 0;
        const float* __restrict__ arow = A + (size_t)rowA * K + kA;

        v8f acc0 = {}, acc1 = {};
        for (int k0 = 0; k0 < K; k0 += 32) {
            // A fragment: one contiguous 16-float run -> 4x b128 loads
            const float4* ap = (const float4*)(arow + k0);
            const float4 a0 = ap[0], a1 = ap[1];   // K = k0+kA .. +7
            const float4 a2 = ap[2], a3 = ap[3];   // K = k0+kA+8 .. +15
            v16h a;
            a[0]  = (_Float16)a0.x; a[1]  = (_Float16)a0.y;
            a[2]  = (_Float16)a0.z; a[3]  = (_Float16)a0.w;
            a[4]  = (_Float16)a1.x; a[5]  = (_Float16)a1.y;
            a[6]  = (_Float16)a1.z; a[7]  = (_Float16)a1.w;
            a[8]  = (_Float16)a2.x; a[9]  = (_Float16)a2.y;
            a[10] = (_Float16)a2.z; a[11] = (_Float16)a2.w;
            a[12] = (_Float16)a3.x; a[13] = (_Float16)a3.y;
            a[14] = (_Float16)a3.z; a[15] = (_Float16)a3.w;

            // B fragments from LDS (contiguous 32B per lane)
            const int kb   = k0 + (hi ? 16 : 0);
            const int base = ((kb >> 4) << 9);
            const v16h b0 = *(const v16h*)&b_sh[base + (l16 << 4)];
            const v16h b1 = *(const v16h*)&b_sh[base + ((l16 + 16) << 4)];

            acc0 = __builtin_amdgcn_wmma_f32_16x16x32_f16(
                false, a, false, b0, (short)0, acc0, false, false);
            acc1 = __builtin_amdgcn_wmma_f32_16x16x32_f16(
                false, a, false, b1, (short)0, acc1, false, false);
        }

        const int colN0 = tileN + l16;
        const int colN1 = colN0 + 16;
        const float bv0 = bias ? bias[colN0] : 0.0f;
        const float bv1 = bias ? bias[colN1] : 0.0f;
        const int rbase = tileM + (hi ? 8 : 0);
#pragma unroll
        for (int r = 0; r < 8; ++r) {
            const size_t rowoff = (size_t)(rbase + r) * Nc;
            float v0 = acc0[r] + bv0;
            float v1 = acc1[r] + bv1;
            if (SILU) { v0 = silu_f(v0); v1 = silu_f(v1); }
            C[rowoff + colN0] = v0;
            C[rowoff + colN1] = v1;
        }
    }
}

// ---------------------------------------------------------------------------
// s = embed_W[atoms]
// ---------------------------------------------------------------------------
__global__ void embed_kernel(const int* __restrict__ atoms,
                             const float* __restrict__ embed_W,
                             float* __restrict__ s, int N) {
    int idx = blockIdx.x * blockDim.x + threadIdx.x;
    if (idx >= N * F_DIM) return;
    int n = idx / F_DIM, f = idx - n * F_DIM;
    s[idx] = embed_W[(size_t)atoms[n] * F_DIM + f];
}

__global__ void zero_kernel(float* __restrict__ p, int n) {
    int idx = blockIdx.x * blockDim.x + threadIdx.x;
    if (idx < n) p[idx] = 0.0f;
}

// ---------------------------------------------------------------------------
// Per-edge geometry: unit dir u[E,3], rbf[E,NRBF], fcut[E]
// ---------------------------------------------------------------------------
__global__ void edge_geom_kernel(const float* __restrict__ pos,
                                 const int* __restrict__ esrc,
                                 const int* __restrict__ edst,
                                 float* __restrict__ u_e,
                                 float* __restrict__ rbf_e,
                                 float* __restrict__ fcut_e, int E) {
    int e = blockIdx.x * blockDim.x + threadIdx.x;
    if (e >= E) return;
    int s = esrc[e], d = edst[e];
    float rx = pos[3 * s + 0] - pos[3 * d + 0];
    float ry = pos[3 * s + 1] - pos[3 * d + 1];
    float rz = pos[3 * s + 2] - pos[3 * d + 2];
    float dist = sqrtf(rx * rx + ry * ry + rz * rz);
    float inv = 1.0f / dist;
    u_e[3 * e + 0] = rx * inv;
    u_e[3 * e + 1] = ry * inv;
    u_e[3 * e + 2] = rz * inv;
    const float c = (float)M_PI * dist / RC;
#pragma unroll
    for (int k = 0; k < NRBF; ++k)
        rbf_e[(size_t)e * NRBF + k] = sinf((float)(k + 1) * c) * inv;
    fcut_e[e] = 0.5f * (cosf(c) + 1.0f);
}

// ---------------------------------------------------------------------------
// Message aggregation. One 128-thread block per node; thread = channel c.
// edge_dst sorted -> binary-search [lo,hi) range; sequential (deterministic)
// reduction over edges. rbf filter weights (20x384 = 30 KB) staged in LDS.
// ---------------------------------------------------------------------------
__device__ __forceinline__ int lower_bound_i(const int* __restrict__ a, int n, int v) {
    int lo = 0, hi = n;
    while (lo < hi) {
        int mid = (lo + hi) >> 1;
        if (a[mid] < v) lo = mid + 1; else hi = mid;
    }
    return lo;
}

__global__ __launch_bounds__(128)
void msg_agg_kernel(const float* __restrict__ s_in, const float* __restrict__ v_in,
                    const float* __restrict__ phi,        // [N,3F]
                    const float* __restrict__ rbf_w_l,    // [NRBF,3F]
                    const float* __restrict__ rbf_b_l,    // [3F]
                    const float* __restrict__ u_e, const float* __restrict__ rbf_e,
                    const float* __restrict__ fcut_e,
                    const int* __restrict__ esrc, const int* __restrict__ edst,
                    float* __restrict__ s_out, float* __restrict__ v_out,
                    int N, int E) {
    __shared__ float w_sh[NRBF * 3 * F_DIM];   // 30720 B
    __shared__ float b_sh[3 * F_DIM];
    for (int i = threadIdx.x; i < NRBF * 3 * F_DIM; i += 128) w_sh[i] = rbf_w_l[i];
    for (int i = threadIdx.x; i < 3 * F_DIM; i += 128) b_sh[i] = rbf_b_l[i];
    __syncthreads();

    const int node = blockIdx.x;
    if (node >= N) return;
    const int c = threadIdx.x;
    const int lo = lower_bound_i(edst, E, node);
    const int hi = lower_bound_i(edst, E, node + 1);

    float ds = 0.0f, dv0 = 0.0f, dv1 = 0.0f, dv2 = 0.0f;
    for (int e = lo; e < hi; ++e) {
        const int src = esrc[e];
        const float fc = fcut_e[e];
        float wf_s = b_sh[c], wf_vv = b_sh[F_DIM + c], wf_vu = b_sh[2 * F_DIM + c];
#pragma unroll
        for (int k = 0; k < NRBF; ++k) {
            const float rb = rbf_e[(size_t)e * NRBF + k];
            wf_s  += rb * w_sh[k * 3 * F_DIM + c];
            wf_vv += rb * w_sh[k * 3 * F_DIM + F_DIM + c];
            wf_vu += rb * w_sh[k * 3 * F_DIM + 2 * F_DIM + c];
        }
        const size_t pb = (size_t)src * 3 * F_DIM;
        const float xs  = phi[pb + c]              * wf_s  * fc;
        const float xvv = phi[pb + F_DIM + c]      * wf_vv * fc;
        const float xvu = phi[pb + 2 * F_DIM + c]  * wf_vu * fc;
        ds += xs;
        dv0 += xvv * v_in[pb + 0 * F_DIM + c] + xvu * u_e[3 * e + 0];
        dv1 += xvv * v_in[pb + 1 * F_DIM + c] + xvu * u_e[3 * e + 1];
        dv2 += xvv * v_in[pb + 2 * F_DIM + c] + xvu * u_e[3 * e + 2];
    }
    const size_t sb = (size_t)node * F_DIM;
    const size_t vb = (size_t)node * 3 * F_DIM;
    s_out[sb + c] = s_in[sb + c] + ds;
    v_out[vb + 0 * F_DIM + c] = v_in[vb + 0 * F_DIM + c] + dv0;
    v_out[vb + 1 * F_DIM + c] = v_in[vb + 1 * F_DIM + c] + dv1;
    v_out[vb + 2 * F_DIM + c] = v_in[vb + 2 * F_DIM + c] + dv2;
}

// ---------------------------------------------------------------------------
// cat[n, 0:F] = s ; cat[n, F:2F] = sqrt(sum_i Vv^2 + 1e-8)
// ---------------------------------------------------------------------------
__global__ __launch_bounds__(128)
void cat_vn_kernel(const float* __restrict__ s, const float* __restrict__ Vv,
                   float* __restrict__ cat, int N) {
    const int n = blockIdx.x;
    if (n >= N) return;
    const int c = threadIdx.x;
    const size_t vb = (size_t)n * 3 * F_DIM;
    const float a0 = Vv[vb + 0 * F_DIM + c];
    const float a1 = Vv[vb + 1 * F_DIM + c];
    const float a2 = Vv[vb + 2 * F_DIM + c];
    cat[(size_t)n * 2 * F_DIM + c]         = s[(size_t)n * F_DIM + c];
    cat[(size_t)n * 2 * F_DIM + F_DIM + c] = sqrtf(a0 * a0 + a1 * a1 + a2 * a2 + 1e-8f);
}

// ---------------------------------------------------------------------------
// Update apply: v_out = v + avv*Uv ; s_out = s + asv*<Uv,Vv> + ass
// ---------------------------------------------------------------------------
__global__ __launch_bounds__(128)
void update_apply_kernel(const float* __restrict__ s_in, const float* __restrict__ v_in,
                         const float* __restrict__ Uv, const float* __restrict__ Vv,
                         const float* __restrict__ a,  // [N,3F]
                         float* __restrict__ s_out, float* __restrict__ v_out, int N) {
    const int n = blockIdx.x;
    if (n >= N) return;
    const int c = threadIdx.x;
    const size_t vb = (size_t)n * 3 * F_DIM;
    const float u0 = Uv[vb + 0 * F_DIM + c], w0 = Vv[vb + 0 * F_DIM + c];
    const float u1 = Uv[vb + 1 * F_DIM + c], w1 = Vv[vb + 1 * F_DIM + c];
    const float u2 = Uv[vb + 2 * F_DIM + c], w2 = Vv[vb + 2 * F_DIM + c];
    const float avv = a[vb + c];
    const float asv = a[vb + F_DIM + c];
    const float ass = a[vb + 2 * F_DIM + c];
    v_out[vb + 0 * F_DIM + c] = v_in[vb + 0 * F_DIM + c] + avv * u0;
    v_out[vb + 1 * F_DIM + c] = v_in[vb + 1 * F_DIM + c] + avv * u1;
    v_out[vb + 2 * F_DIM + c] = v_in[vb + 2 * F_DIM + c] + avv * u2;
    s_out[(size_t)n * F_DIM + c] =
        s_in[(size_t)n * F_DIM + c] + asv * (u0 * w0 + u1 * w1 + u2 * w2) + ass;
}

// ---------------------------------------------------------------------------
// Readout: out[n] = sum_c t[n,c]*out_w2[c] + out_b2  (NOUT = 1)
// ---------------------------------------------------------------------------
__global__ __launch_bounds__(128)
void readout_kernel(const float* __restrict__ t, const float* __restrict__ out_w2,
                    const float* __restrict__ out_b2, float* __restrict__ out, int N) {
    __shared__ float red[128];
    const int n = blockIdx.x;
    if (n >= N) return;
    const int c = threadIdx.x;
    red[c] = t[(size_t)n * F_DIM + c] * out_w2[c];
    __syncthreads();
#pragma unroll
    for (int s = 64; s > 0; s >>= 1) {
        if (c < s) red[c] += red[c + s];
        __syncthreads();
    }
    if (c == 0) out[n] = red[0] + out_b2[0];
}

// ---------------------------------------------------------------------------
// Host-side orchestration
// ---------------------------------------------------------------------------
static inline dim3 gemm_grid(int M, int Nc) {
    return dim3((unsigned)(Nc / 32), (unsigned)((M + 63) / 64), 1);
}

extern "C" void kernel_launch(void* const* d_in, const int* in_sizes, int n_in,
                              void* d_out, int out_size, void* d_ws, size_t ws_size,
                              hipStream_t stream) {
    const int*   atoms   = (const int*)d_in[0];
    const float* pos     = (const float*)d_in[1];
    // d_in[2] = graph_indexes (unused: cutoff never crosses molecules)
    const int*   esrc    = (const int*)d_in[3];
    const int*   edst    = (const int*)d_in[4];
    const float* embed_W = (const float*)d_in[5];
    const float* msg_w1  = (const float*)d_in[6];
    const float* msg_b1  = (const float*)d_in[7];
    const float* msg_w2  = (const float*)d_in[8];
    const float* msg_b2  = (const float*)d_in[9];
    const float* rbf_w   = (const float*)d_in[10];
    const float* rbf_b   = (const float*)d_in[11];
    const float* upd_U   = (const float*)d_in[12];
    const float* upd_V   = (const float*)d_in[13];
    const float* upd_w1  = (const float*)d_in[14];
    const float* upd_b1  = (const float*)d_in[15];
    const float* upd_w2  = (const float*)d_in[16];
    const float* upd_b2  = (const float*)d_in[17];
    const float* out_w1  = (const float*)d_in[18];
    const float* out_b1  = (const float*)d_in[19];
    const float* out_w2  = (const float*)d_in[20];
    const float* out_b2  = (const float*)d_in[21];

    const int N = in_sizes[0];
    const int E = in_sizes[3];
    const int F = F_DIM;

    // workspace carve-up (fp32)
    size_t off = 0;
    auto carve = [&](size_t nfl) {
        float* p = (float*)((char*)d_ws + off);
        off += nfl * sizeof(float);
        return p;
    };
    float* s_a  = carve((size_t)N * F);
    float* s_b  = carve((size_t)N * F);
    float* v_a  = carve((size_t)N * 3 * F);
    float* v_b  = carve((size_t)N * 3 * F);
    float* h1   = carve((size_t)N * F);        // MLP hidden (shared)
    float* phi  = carve((size_t)N * 3 * F);    // also reused for 'a'
    float* Uv   = carve((size_t)N * 3 * F);
    float* Vv   = carve((size_t)N * 3 * F);
    float* cat  = carve((size_t)N * 2 * F);
    float* u_e  = carve((size_t)E * 3);
    float* rbfe = carve((size_t)E * NRBF);
    float* fce  = carve((size_t)E);
    (void)ws_size;

    // init: s = embed_W[atoms], v = 0
    embed_kernel<<<(N * F + 255) / 256, 256, 0, stream>>>(atoms, embed_W, s_a, N);
    zero_kernel<<<(N * 3 * F + 255) / 256, 256, 0, stream>>>(v_a, N * 3 * F);
    edge_geom_kernel<<<(E + 255) / 256, 256, 0, stream>>>(pos, esrc, edst, u_e, rbfe, fce, E);

    float* s_cur = s_a; float* v_cur = v_a;
    float* s_nxt = s_b; float* v_nxt = v_b;

    for (int l = 0; l < NLAYERS; ++l) {
        const float* mw1 = msg_w1 + (size_t)l * F * F;
        const float* mb1 = msg_b1 + (size_t)l * F;
        const float* mw2 = msg_w2 + (size_t)l * F * 3 * F;
        const float* mb2 = msg_b2 + (size_t)l * 3 * F;
        const float* rw  = rbf_w  + (size_t)l * NRBF * 3 * F;
        const float* rb  = rbf_b  + (size_t)l * 3 * F;
        const float* uU  = upd_U  + (size_t)l * F * F;
        const float* uV  = upd_V  + (size_t)l * F * F;
        const float* uw1 = upd_w1 + (size_t)l * 2 * F * F;
        const float* ub1 = upd_b1 + (size_t)l * F;
        const float* uw2 = upd_w2 + (size_t)l * F * 3 * F;
        const float* ub2 = upd_b2 + (size_t)l * 3 * F;

        // ---- message block ----
        wmma_gemm_kernel<1><<<gemm_grid(N, F), 128, 0, stream>>>(s_cur, mw1, mb1, h1, N, F, F);
        wmma_gemm_kernel<0><<<gemm_grid(N, 3 * F), 128, 0, stream>>>(h1, mw2, mb2, phi, N, F, 3 * F);
        msg_agg_kernel<<<N, 128, 0, stream>>>(s_cur, v_cur, phi, rw, rb,
                                              u_e, rbfe, fce, esrc, edst,
                                              s_nxt, v_nxt, N, E);
        // ---- update block ----
        wmma_gemm_kernel<0><<<gemm_grid(3 * N, F), 128, 0, stream>>>(v_nxt, uU, nullptr, Uv, 3 * N, F, F);
        wmma_gemm_kernel<0><<<gemm_grid(3 * N, F), 128, 0, stream>>>(v_nxt, uV, nullptr, Vv, 3 * N, F, F);
        cat_vn_kernel<<<N, 128, 0, stream>>>(s_nxt, Vv, cat, N);
        wmma_gemm_kernel<1><<<gemm_grid(N, F), 128, 0, stream>>>(cat, uw1, ub1, h1, N, 2 * F, F);
        wmma_gemm_kernel<0><<<gemm_grid(N, 3 * F), 128, 0, stream>>>(h1, uw2, ub2, phi, N, F, 3 * F);
        update_apply_kernel<<<N, 128, 0, stream>>>(s_nxt, v_nxt, Uv, Vv, phi,
                                                   s_cur, v_cur, N);
        // results now back in s_cur / v_cur
    }

    // ---- readout ----
    wmma_gemm_kernel<1><<<gemm_grid(N, F), 128, 0, stream>>>(s_cur, out_w1, out_b1, h1, N, F, F);
    readout_kernel<<<N, 128, 0, stream>>>(h1, out_w2, out_b2, (float*)d_out, N);
    (void)out_size; (void)n_in;
}
